// _ECELoss_24111946400318
// MI455X (gfx1250) — compile-verified, matrix operand order checked by reading, not run on hardware
//
#include <hip/hip_runtime.h>
#include <hip/hip_bf16.h>
#include <stdint.h>

typedef __attribute__((ext_vector_type(16))) _Float16 v16h;
typedef __attribute__((ext_vector_type(8)))  float    v8f;

#define TPB              256
#define WAVES_PER_BLOCK  8
#define NBINS            15
#define CCOLS            100
#define PADF             104      // padded row stride in floats (416 B, 16B-aligned)
#define ROWS_PER_TILE    16
#define LOG2E            1.44269504088896340736f

// ---------------------------------------------------------------------------
// Zero the 45 accumulator doubles in workspace (must run every launch: the
// harness does not re-poison, and we accumulate with atomics).
// ---------------------------------------------------------------------------
__global__ void ece_zero_kernel(double* __restrict__ bins) {
    int t = threadIdx.x;
    if (t < 3 * NBINS) bins[t] = 0.0;
}

// ---------------------------------------------------------------------------
// Main kernel: wave-per-16-row-tile streaming reduction.
//   HBM read once via async global->LDS (ASYNCcnt), two LDS passes:
//   pass 1 = vectorized row max, pass 2 = exp + WMMA row-sum on matrix pipe.
// ---------------------------------------------------------------------------
__global__ void __launch_bounds__(TPB)
ece_main_kernel(const float* __restrict__ logits,
                const int*   __restrict__ labels,
                double*      __restrict__ bins,
                int N, int nTiles) {
    __shared__ __align__(16) float tile[WAVES_PER_BLOCK][ROWS_PER_TILE * PADF];
    __shared__ float  rowsum[WAVES_PER_BLOCK][ROWS_PER_TILE];
    __shared__ double sb[3 * NBINS];

    const int tid  = threadIdx.x;
    const int lane = tid & 31;
    const int w    = tid >> 5;

    if (tid < 3 * NBINS) sb[tid] = 0.0;
    __syncthreads();

    const int gw = blockIdx.x * WAVES_PER_BLOCK + w;   // global wave id
    const int nw = gridDim.x * WAVES_PER_BLOCK;        // total waves

    // Workgroup-relative LDS byte offset of this wave's tile (generic LDS
    // pointers carry the LDS offset in their low 32 bits).
    const uint32_t ldsBase = (uint32_t)(uintptr_t)(&tile[w][0]);

    const int row16 = lane & 15;   // row handled by this lane (pairs of lanes)
    const int hi    = lane >> 4;   // which half of the lane pair

    // B matrix = all ones (32x16 f16): layout-independent because uniform.
    v16h ones;
#pragma unroll
    for (int i = 0; i < 16; ++i) ones[i] = (_Float16)1.0f;

    for (int t = gw; t < nTiles; t += nw) {
        const float* gbase = logits + (size_t)t * ROWS_PER_TILE * CCOLS;

        // ------------------------------------------------------------------
        // Stage 16x100 floats into LDS: 400 x 16B segments on the CDNA5
        // async global->LDS path. Drain previous-iteration LDS reads first.
        // ------------------------------------------------------------------
        asm volatile("s_wait_dscnt 0" ::: "memory");
#pragma unroll
        for (int it = 0; it < 13; ++it) {
            int s = it * 32 + lane;
            if (s < 400) {
                int r   = s / 25;
                int seg = s % 25;
                if (t * ROWS_PER_TILE + r < N) {
                    const float* gp = gbase + r * CCOLS + seg * 4;
                    uint32_t lo = ldsBase + (uint32_t)(r * (PADF * 4) + seg * 16);
                    asm volatile("global_load_async_to_lds_b128 %0, %1, off"
                                 :: "v"(lo), "v"(gp) : "memory");
                }
            }
        }
        asm volatile("s_wait_asynccnt 0x0" ::: "memory");

        // ------------------------------------------------------------------
        // Pass 1: row max only (prediction check is done via the label's
        // logit, so no argmax index tracking). Lane pair splits the row's
        // 25 float4 segments 13/12; merge across the pair with shfl_xor.
        // ------------------------------------------------------------------
        const float* rp = &tile[w][row16 * PADF];
        const float4* rp4 = reinterpret_cast<const float4*>(rp);
        float m = -3.402823466e38f;
        {
            int s0 = hi ? 13 : 0;
            int s1 = hi ? 25 : 13;
#pragma unroll 13
            for (int s = s0; s < s1; ++s) {
                float4 v = rp4[s];
                m = fmaxf(m, fmaxf(fmaxf(v.x, v.y), fmaxf(v.z, v.w)));
            }
        }
        m = fmaxf(m, __shfl_xor(m, 16));
        const float mlg = m * LOG2E;   // exp(v-m) = exp2(v*log2e - m*log2e)

        // ------------------------------------------------------------------
        // Pass 2: exp + row-sum via 4x v_wmma_f32_16x16x32_f16 against a
        // ones matrix (K padded 100 -> 128 with zeros). A-fragment layout
        // (16-bit A 16x32): lane l, VGPR v holds
        //   K = (v<4 ? 2v : 16+2(v-4)) + (l>=16 ? 8 : 0), K+1   for row l&15.
        // ------------------------------------------------------------------
        v8f acc = {};
#pragma unroll
        for (int ch = 0; ch < 4; ++ch) {
            v16h a;
#pragma unroll
            for (int v = 0; v < 8; ++v) {
                int k = ch * 32 + (v < 4 ? 2 * v : 16 + 2 * (v - 4)) + hi * 8;
                float e0 = 0.0f, e1 = 0.0f;
                if (k     < CCOLS) e0 = __builtin_amdgcn_exp2f(fmaf(rp[k],     LOG2E, -mlg));
                if (k + 1 < CCOLS) e1 = __builtin_amdgcn_exp2f(fmaf(rp[k + 1], LOG2E, -mlg));
                a[2 * v]     = (_Float16)e0;
                a[2 * v + 1] = (_Float16)e1;
            }
            acc = __builtin_amdgcn_wmma_f32_16x16x32_f16(
                      false, a, false, ones, (short)0, acc, false, false);
        }

        // D layout: VGPR r -> row r (lanes 0-15) / row 8+r (lanes 16-31).
        if (lane == 0) {
#pragma unroll
            for (int r = 0; r < 8; ++r) rowsum[w][r] = acc[r];
        }
        if (lane == 16) {
#pragma unroll
            for (int r = 0; r < 8; ++r) rowsum[w][8 + r] = acc[r];
        }
        // LDS ops are in-order within a wave; reads below see the stores.

        // ------------------------------------------------------------------
        // Finalize: one lane per row -> confidence, bin, atomically bin it.
        // prediction==label  <=>  logits[row][label] == rowmax.
        // ------------------------------------------------------------------
        if (hi == 0) {
            int grow = t * ROWS_PER_TILE + row16;
            if (grow < N) {
                float s    = rowsum[w][row16];
                float conf = 1.0f / s;                 // exp(0)=1 term included
                int   lab  = labels[grow];
                float accv = (rp[lab] == m) ? 1.0f : 0.0f;
                int bin = (int)ceilf(conf * (float)NBINS) - 1;
                bin = bin < 0 ? 0 : (bin > NBINS - 1 ? NBINS - 1 : bin);
                atomicAdd(&sb[bin],              1.0);
                atomicAdd(&sb[NBINS + bin],      (double)conf);
                atomicAdd(&sb[2 * NBINS + bin],  (double)accv);
            }
        }
    }

    __syncthreads();
    if (tid < 3 * NBINS) atomicAdd(&bins[tid], sb[tid]);
}

// ---------------------------------------------------------------------------
// Epilogue: 15-bin ECE reduction, single thread (trivial work).
// ---------------------------------------------------------------------------
__global__ void ece_final_kernel(const double* __restrict__ bins,
                                 float* __restrict__ out, int N) {
    if (threadIdx.x == 0 && blockIdx.x == 0) {
        double invN = 1.0 / (double)N;
        double e = 0.0;
        for (int b = 0; b < NBINS; ++b) {
            double cnt = bins[b];
            if (cnt > 0.0) {
                double conf = bins[NBINS + b] / cnt;
                double acc  = bins[2 * NBINS + b] / cnt;
                e += fabs(conf - acc) * cnt * invN;
            }
        }
        out[0] = (float)e;
    }
}

// ---------------------------------------------------------------------------
extern "C" void kernel_launch(void* const* d_in, const int* in_sizes, int n_in,
                              void* d_out, int out_size, void* d_ws, size_t ws_size,
                              hipStream_t stream) {
    const float* logits = (const float*)d_in[0];
    const int*   labels = (const int*)d_in[1];
    int N = in_sizes[1];                              // number of rows
    int nTiles = (N + ROWS_PER_TILE - 1) / ROWS_PER_TILE;
    double* bins = (double*)d_ws;                     // 45 doubles of scratch

    ece_zero_kernel<<<1, 64, 0, stream>>>(bins);

    int blocks = 2048;                                // 16384 waves, grid-stride
    if (blocks > nTiles) blocks = (nTiles + WAVES_PER_BLOCK - 1) / WAVES_PER_BLOCK;
    if (blocks < 1) blocks = 1;
    ece_main_kernel<<<blocks, TPB, 0, stream>>>(logits, labels, bins, N, nTiles);

    ece_final_kernel<<<1, 32, 0, stream>>>(bins, (float*)d_out, N);
}
